// SGlayer_9723805958288
// MI455X (gfx1250) — matrix operand order repaired and checked by reading, not compiled, not run on hardware
//
#include <hip/hip_runtime.h>
#include <hip/hip_bf16.h>
#include <math.h>

// ---------------------------------------------------------------------------
// Problem constants (from the reference)
// ---------------------------------------------------------------------------
constexpr int NN   = 30000;
constexpr int EE   = 480000;
constexpr int HID  = 128;
constexpr int KW   = 500;
constexpr int WDIM = 300;

// ---------------------------------------------------------------------------
// WMMA helpers (CDNA5 gfx1250, wave32, V_WMMA_F32_16X16X32_BF16)
// ---------------------------------------------------------------------------
typedef __attribute__((ext_vector_type(16))) __bf16 bf16x16;
typedef __attribute__((ext_vector_type(8)))  float  floatx8;

#define WMMA_BF16(acc, a, b)                                                   \
    (acc) = __builtin_amdgcn_wmma_f32_16x16x32_bf16(false, (a), false, (b),    \
                                                    (short)0, (acc), false, false)

// A-matrix 16x32 bf16 lane layout: lane m = lane&15, grp = lane>>4.
// Per lane: K chunk [k0+8*grp, +8) -> a[0..7], [k0+16+8*grp, +8) -> a[8..15].
__device__ __forceinline__ void load_a_lds(bf16x16& a, const __bf16* arow,
                                           int k0, int grp) {
    const int ka = k0 + grp * 8;
    const int kb = k0 + 16 + grp * 8;
#pragma unroll
    for (int i = 0; i < 8; ++i) {
        a[i]     = arow[ka + i];
        a[8 + i] = arow[kb + i];
    }
}

// B-matrix 32x16 bf16 lane layout: column n0 + (lane&15); element e -> K row
// k0+8*grp+e (e<8), k0+16+8*grp+(e-8) (e>=8), matching the A layout.
template <bool GUARD>
__device__ __forceinline__ void load_b_glb(bf16x16& b, const float* B, int ldb,
                                           int krow, int n0, int K) {
    const int kb0 = krow & ~31, g = (krow >> 4) & 1, col = n0 + (krow & 15);
#pragma unroll
    for (int e = 0; e < 16; ++e) {
        const int k = kb0 + (e < 8 ? g * 8 + e : 16 + g * 8 + (e - 8));
        b[e] = (!GUARD || k < K) ? (__bf16)B[(size_t)k * ldb + col] : (__bf16)0.0f;
    }
}

// ---------------------------------------------------------------------------
// 32-row-tile GEMM: C[M x 128] = act(A[M x K] @ B[K x 128] + bias (+R))
// block = 256 (8 waves); wave w -> cols [16w,16w+16) for BOTH 16-row subtiles,
// so each B fragment feeds two WMMAs. K is a template constant -> fully
// unrolled WMMA chain, no per-iteration guards when K % 32 == 0.
// In-place A==C is safe (A tile fully staged in LDS before any store).
// ---------------------------------------------------------------------------
template <int K, bool GUARD>
__global__ __launch_bounds__(256) void gemm32(const float* __restrict__ A, int lda,
                                              const float* __restrict__ B, int ldb,
                                              const float* __restrict__ bias,
                                              const float* __restrict__ R,
                                              float* __restrict__ C,
                                              int M, int relu) {
    constexpr int Kp = (K + 31) & ~31;
    constexpr int AP = Kp + 8;
    __shared__ __bf16 As[32 * AP];

    const int tid = threadIdx.x;
    const int m0  = blockIdx.x * 32;

    for (int idx = tid; idx < 32 * Kp; idx += 256) {
        const int r = idx / Kp, c = idx - r * Kp;
        const int m = m0 + r;
        const float v = (m < M && c < K) ? A[(size_t)m * lda + c] : 0.0f;
        As[r * AP + c] = (__bf16)v;
    }
    __syncthreads();

    const int lane = tid & 31, wave = tid >> 5;
    const int grp = lane >> 4, m16 = lane & 15;
    const int n0 = wave * 16;
    floatx8 acc0 = {}, acc1 = {};
    const __bf16* arow0 = &As[m16 * AP];
    const __bf16* arow1 = &As[(16 + m16) * AP];

#pragma unroll
    for (int k0 = 0; k0 < Kp; k0 += 32) {
        bf16x16 a0, a1, b;
        load_b_glb<GUARD>(b, B, ldb, k0 + grp * 16 + m16, n0, K);
        load_a_lds(a0, arow0, k0, grp);
        load_a_lds(a1, arow1, k0, grp);
        WMMA_BF16(acc0, a0, b);
        WMMA_BF16(acc1, a1, b);
    }

    const int n = n0 + m16;
    const float bv = bias ? bias[n] : 0.0f;
#pragma unroll
    for (int r = 0; r < 8; ++r) {
        const int m = m0 + r + grp * 8;
        if (m < M) {
            float v = acc0[r] + bv;
            if (R) v += R[(size_t)m * HID + n];
            if (relu) v = fmaxf(v, 0.0f);
            C[(size_t)m * HID + n] = v;
        }
        const int m2 = m0 + 16 + r + grp * 8;
        if (m2 < M) {
            float v = acc1[r] + bv;
            if (R) v += R[(size_t)m2 * HID + n];
            if (relu) v = fmaxf(v, 0.0f);
            C[(size_t)m2 * HID + n] = v;
        }
    }
}

// ---------------------------------------------------------------------------
// Fused FFN (in-place, 32-row tile): X = X + relu(X@W1 + b1) @ W2 + b2.
// block = 512 (16 waves). Hidden (32x256) lives only in LDS (bf16).
// ---------------------------------------------------------------------------
__global__ __launch_bounds__(512) void ffn32(float* __restrict__ X,
                                             const float* __restrict__ W1,
                                             const float* __restrict__ b1,
                                             const float* __restrict__ W2,
                                             const float* __restrict__ b2,
                                             int M) {
    __shared__ __bf16 Xs[32 * 136];
    __shared__ __bf16 Hs[32 * 264];
    const int tid = threadIdx.x;
    const int m0  = blockIdx.x * 32;

    for (int idx = tid; idx < 32 * HID; idx += 512) {
        const int r = idx >> 7, c = idx & 127;
        const int m = m0 + r;
        Xs[r * 136 + c] = (__bf16)((m < M) ? X[(size_t)m * HID + c] : 0.0f);
    }
    __syncthreads();

    const int lane = tid & 31, wave = tid >> 5;
    const int grp = lane >> 4, m16 = lane & 15;

    {   // phase 1: H = relu(X @ W1 + b1); 16 waves x 16 cols, 2 row subtiles
        const int n0 = wave * 16;
        floatx8 acc0 = {}, acc1 = {};
        const __bf16* arow0 = &Xs[m16 * 136];
        const __bf16* arow1 = &Xs[(16 + m16) * 136];
#pragma unroll
        for (int k0 = 0; k0 < HID; k0 += 32) {
            bf16x16 a0, a1, b;
            load_b_glb<false>(b, W1, 2 * HID, k0 + grp * 16 + m16, n0, HID);
            load_a_lds(a0, arow0, k0, grp);
            load_a_lds(a1, arow1, k0, grp);
            WMMA_BF16(acc0, a0, b);
            WMMA_BF16(acc1, a1, b);
        }
        const int n = n0 + m16;
        const float bv = b1 ? b1[n] : 0.0f;
#pragma unroll
        for (int r = 0; r < 8; ++r) {
            Hs[(r + grp * 8) * 264 + n]      = (__bf16)fmaxf(acc0[r] + bv, 0.0f);
            Hs[(16 + r + grp * 8) * 264 + n] = (__bf16)fmaxf(acc1[r] + bv, 0.0f);
        }
    }
    __syncthreads();

    if (wave < 8) {   // phase 2: X += H @ W2 + b2
        const int n0 = wave * 16;
        floatx8 acc0 = {}, acc1 = {};
        const __bf16* arow0 = &Hs[m16 * 264];
        const __bf16* arow1 = &Hs[(16 + m16) * 264];
#pragma unroll
        for (int k0 = 0; k0 < 2 * HID; k0 += 32) {
            bf16x16 a0, a1, b;
            load_b_glb<false>(b, W2, HID, k0 + grp * 16 + m16, n0, 2 * HID);
            load_a_lds(a0, arow0, k0, grp);
            load_a_lds(a1, arow1, k0, grp);
            WMMA_BF16(acc0, a0, b);
            WMMA_BF16(acc1, a1, b);
        }
        const int n = n0 + m16;
        const float bv = b2 ? b2[n] : 0.0f;
#pragma unroll
        for (int r = 0; r < 8; ++r) {
            const int m = m0 + r + grp * 8;
            if (m < M) {
                const size_t o = (size_t)m * HID + n;
                X[o] = acc0[r] + bv + X[o];
            }
            const int m2 = m0 + 16 + r + grp * 8;
            if (m2 < M) {
                const size_t o = (size_t)m2 * HID + n;
                X[o] = acc1[r] + bv + X[o];
            }
        }
    }
}

// ---------------------------------------------------------------------------
// In-place LayerNorm over rows of 128. block = 128, grid = rows.
// ---------------------------------------------------------------------------
__global__ __launch_bounds__(128) void ln128(float* __restrict__ X,
                                             const float* __restrict__ g,
                                             const float* __restrict__ b) {
    __shared__ float red[128];
    const size_t row = blockIdx.x;
    const int t = threadIdx.x;
    const float x = X[row * HID + t];
    red[t] = x;  __syncthreads();
    for (int s = 64; s > 0; s >>= 1) { if (t < s) red[t] += red[t + s]; __syncthreads(); }
    const float mean = red[0] / (float)HID;
    __syncthreads();
    const float d = x - mean;
    red[t] = d * d;  __syncthreads();
    for (int s = 64; s > 0; s >>= 1) { if (t < s) red[t] += red[t + s]; __syncthreads(); }
    const float var = red[0] / (float)HID;
    X[row * HID + t] = d * rsqrtf(var + 1e-5f) * g[t] + b[t];
}

// ---------------------------------------------------------------------------
// Edge attention: score = K[src]*Q[dst]/4 * pe (overwrites pe => e_out),
// s = exp(clip(sum_d score)), atomic segment sums wV += V[src]*s, z += s.
// block = 256 (two edges), half-wave shfl_xor reduction over D=16.
// ---------------------------------------------------------------------------
__global__ __launch_bounds__(256) void edge_attn(const float* __restrict__ Q,
                                                 const float* __restrict__ Kf,
                                                 const float* __restrict__ V,
                                                 float* __restrict__ PE,
                                                 const int* __restrict__ src,
                                                 const int* __restrict__ dst,
                                                 float* __restrict__ wV,
                                                 float* __restrict__ z) {
    const int t = threadIdx.x & 127;
    const size_t e = (size_t)blockIdx.x * 2 + (threadIdx.x >> 7);
    if (e >= EE) return;
    const int s_i = src[e], d_i = dst[e];
    const float q = Q[(size_t)d_i * HID + t];
    const float k = Kf[(size_t)s_i * HID + t];
    const float p = PE[e * HID + t];
    const float sc = k * q * 0.25f * p;
    PE[e * HID + t] = sc;                       // e_out
    float sum = sc;
    sum += __shfl_xor(sum, 1);
    sum += __shfl_xor(sum, 2);
    sum += __shfl_xor(sum, 4);
    sum += __shfl_xor(sum, 8);                  // full D=16 head sum
    const float s = expf(fminf(fmaxf(sum, -5.0f), 5.0f));
    atomicAdd(&wV[(size_t)d_i * HID + t], V[(size_t)s_i * HID + t] * s);
    if ((t & 15) == 0) atomicAdd(&z[(size_t)d_i * 8 + (t >> 4)], s);
}

__global__ __launch_bounds__(256) void finalize_hatt(const float* __restrict__ wV,
                                                     const float* __restrict__ z,
                                                     float* __restrict__ hatt) {
    const size_t idx = (size_t)blockIdx.x * 256 + threadIdx.x;
    if (idx < (size_t)NN * HID) {
        const size_t n = idx >> 7;
        const int h = (int)((idx >> 4) & 7);
        hatt[idx] = wV[idx] / (z[n * 8 + h] + 1e-6f);
    }
}

// kT[c][r] = k[r][c], zero-padded rows KW..511. kT is 128 x 512.
__global__ __launch_bounds__(256) void transpose_kT(const float* __restrict__ k,
                                                    float* __restrict__ kT) {
    const int idx = blockIdx.x * 256 + threadIdx.x;
    if (idx < 128 * 512) {
        const int c = idx >> 9;
        const int r = idx & 511;
        kT[idx] = (r < KW) ? k[(size_t)r * HID + c] : 0.0f;
    }
}

// ---------------------------------------------------------------------------
// Cross-attention: logits = softmax(q @ kT / sqrt(128)) [N x 500] (f32 out),
// attv = logits @ v. block = 256 (8 waves), one 16-node tile per block.
// ---------------------------------------------------------------------------
__global__ __launch_bounds__(256) void cross_attn(const float* __restrict__ q,
                                                  const float* __restrict__ kT,
                                                  const float* __restrict__ vv,
                                                  float* __restrict__ logits,
                                                  float* __restrict__ attv) {
    __shared__ __bf16 Qs[16 * 136];
    __shared__ float  S[16 * 520];
    __shared__ __bf16 P[16 * 520];
    const int tid = threadIdx.x;
    const int m0  = blockIdx.x * 16;
    const float scale = 0.0883883476483184f;   // 1/sqrt(128)

    for (int idx = tid; idx < 16 * HID; idx += 256) {
        const int r = idx >> 7, c = idx & 127;
        Qs[r * 136 + c] = (__bf16)(q[(size_t)(m0 + r) * HID + c] * scale);
    }
    __syncthreads();

    const int lane = tid & 31, wave = tid >> 5;
    const int grp = lane >> 4, m16 = lane & 15;
    const __bf16* arow = &Qs[m16 * 136];

#pragma unroll
    for (int pass = 0; pass < 4; ++pass) {     // 512 score cols = 4 x (8 x 16)
        const int n0 = pass * 128 + wave * 16;
        floatx8 acc = {};
#pragma unroll
        for (int k0 = 0; k0 < HID; k0 += 32) {
            bf16x16 a, b;
            load_b_glb<false>(b, kT, 512, k0 + grp * 16 + m16, n0, HID);
            load_a_lds(a, arow, k0, grp);
            WMMA_BF16(acc, a, b);
        }
        const int n = n0 + m16;
#pragma unroll
        for (int r = 0; r < 8; ++r) S[(r + grp * 8) * 520 + n] = acc[r];
    }
    __syncthreads();

    {   // softmax over 500 cols; 16 threads per row, xor-reduce within 16-group
        const int row = tid >> 4, w16 = tid & 15;
        float* srow = &S[row * 520];
        float mx = -1e30f;
        for (int c = w16; c < KW; c += 16) mx = fmaxf(mx, srow[c]);
        mx = fmaxf(mx, __shfl_xor(mx, 1));
        mx = fmaxf(mx, __shfl_xor(mx, 2));
        mx = fmaxf(mx, __shfl_xor(mx, 4));
        mx = fmaxf(mx, __shfl_xor(mx, 8));
        float sm = 0.0f;
        for (int c = w16; c < KW; c += 16) sm += expf(srow[c] - mx);
        sm += __shfl_xor(sm, 1);
        sm += __shfl_xor(sm, 2);
        sm += __shfl_xor(sm, 4);
        sm += __shfl_xor(sm, 8);
        const float inv = 1.0f / sm;
        for (int c = w16; c < KW; c += 16) {
            const float pv = expf(srow[c] - mx) * inv;
            logits[(size_t)(m0 + row) * KW + c] = pv;   // f32 exact output
            P[row * 520 + c] = (__bf16)pv;              // bf16 copy for GEMM
        }
        for (int c = KW + w16; c < 512; c += 16) P[row * 520 + c] = (__bf16)0.0f;
    }
    __syncthreads();

    {   // attv = P @ v  (K padded to 512, v rows guarded at 500)
        const int n0 = wave * 16;
        floatx8 acc = {};
        const __bf16* prow = &P[m16 * 520];
#pragma unroll
        for (int k0 = 0; k0 < 512; k0 += 32) {
            bf16x16 a, b;
            load_b_glb<true>(b, vv, HID, k0 + grp * 16 + m16, n0, KW);
            load_a_lds(a, prow, k0, grp);
            WMMA_BF16(acc, a, b);
        }
        const int n = n0 + m16;
#pragma unroll
        for (int r = 0; r < 8; ++r)
            attv[(size_t)(m0 + r + grp * 8) * HID + n] = acc[r];
    }
}

// ---------------------------------------------------------------------------
// Host-side orchestration
// ---------------------------------------------------------------------------
extern "C" void kernel_launch(void* const* d_in, const int* in_sizes, int n_in,
                              void* d_out, int out_size, void* d_ws, size_t ws_size,
                              hipStream_t stream) {
    (void)in_sizes; (void)n_in; (void)out_size; (void)ws_size;

    const float* node  = (const float*)d_in[0];
    const float* edge  = (const float*)d_in[1];
    const float* word  = (const float*)d_in[2];
    const int*   src   = (const int*)d_in[3];
    const int*   dst   = (const int*)d_in[4];
    const float* Wq    = (const float*)d_in[5];
    const float* Wk    = (const float*)d_in[6];
    const float* Wv    = (const float*)d_in[7];
    const float* We    = (const float*)d_in[8];
    const float* Oh_w  = (const float*)d_in[9];
    const float* Oh_b  = (const float*)d_in[10];
    const float* Oe_w  = (const float*)d_in[11];
    const float* Oe_b  = (const float*)d_in[12];
    const float* ln1h_g = (const float*)d_in[13];
    const float* ln1h_b = (const float*)d_in[14];
    const float* ln1e_g = (const float*)d_in[15];
    const float* ln1e_b = (const float*)d_in[16];
    const float* Fh1_w = (const float*)d_in[17];
    const float* Fh1_b = (const float*)d_in[18];
    const float* Fh2_w = (const float*)d_in[19];
    const float* Fh2_b = (const float*)d_in[20];
    const float* Fe1_w = (const float*)d_in[21];
    const float* Fe1_b = (const float*)d_in[22];
    const float* Fe2_w = (const float*)d_in[23];
    const float* Fe2_b = (const float*)d_in[24];
    const float* ln2h_g = (const float*)d_in[25];
    const float* ln2h_b = (const float*)d_in[26];
    const float* ln2e_g = (const float*)d_in[27];
    const float* ln2e_b = (const float*)d_in[28];
    const float* ca_q   = (const float*)d_in[29];
    const float* ca_k   = (const float*)d_in[30];
    const float* ca_v   = (const float*)d_in[31];
    const float* ca_att = (const float*)d_in[32];
    const float* ca_ln1_g = (const float*)d_in[33];
    const float* ca_ln1_b = (const float*)d_in[34];
    const float* ca_ln2_g = (const float*)d_in[35];
    const float* ca_ln2_b = (const float*)d_in[36];
    const float* ca_f1  = (const float*)d_in[37];
    const float* ca_f2  = (const float*)d_in[38];

    float* out    = (float*)d_out;
    float* logits = out;                               // N*KW
    float* nf     = logits + (size_t)NN * KW;          // N*HID
    float* ef     = nf + (size_t)NN * HID;             // E*HID (pe/e_out live here too)

    float* ws   = (float*)d_ws;
    float* Qb   = ws;
    float* Kb   = Qb + (size_t)NN * HID;
    float* Vb   = Kb + (size_t)NN * HID;
    float* wV   = Vb + (size_t)NN * HID;
    float* zb   = wV + (size_t)NN * HID;
    float* hf   = zb + (size_t)NN * 8;
    float* kT   = hf + (size_t)NN * HID;
    float* vv   = kT + (size_t)128 * 512;
    float* attv = vv + (size_t)KW * HID;

    const int gN32 = (NN + 31) / 32;   // 938
    const int gE32 = (EE + 31) / 32;   // 15000
    const int gW32 = (KW + 31) / 32;   // 16
    const int gN16 = (NN + 15) / 16;   // 1875

    // --- graph attention projections ---
    gemm32<HID, false><<<gN32, 256, 0, stream>>>(node, HID, Wq, HID, nullptr, nullptr, Qb, NN, 0);
    gemm32<HID, false><<<gN32, 256, 0, stream>>>(node, HID, Wk, HID, nullptr, nullptr, Kb, NN, 0);
    gemm32<HID, false><<<gN32, 256, 0, stream>>>(node, HID, Wv, HID, nullptr, nullptr, Vb, NN, 0);
    gemm32<HID, false><<<gE32, 256, 0, stream>>>(edge, HID, We, HID, nullptr, nullptr, ef, EE, 0);

    hipMemsetAsync(wV, 0, (size_t)(NN * HID + NN * 8) * sizeof(float), stream);
    edge_attn<<<(EE + 1) / 2, 256, 0, stream>>>(Qb, Kb, Vb, ef, src, dst, wV, zb);
    finalize_hatt<<<(NN * HID + 255) / 256, 256, 0, stream>>>(wV, zb, Qb);

    // --- SALayer: residual + LN + FFN (nodes & edges) ---
    gemm32<HID, false><<<gN32, 256, 0, stream>>>(Qb, HID, Oh_w, HID, Oh_b, node, hf, NN, 0);
    ln128<<<NN, 128, 0, stream>>>(hf, ln1h_g, ln1h_b);
    gemm32<HID, false><<<gE32, 256, 0, stream>>>(ef, HID, Oe_w, HID, Oe_b, edge, ef, EE, 0);
    ln128<<<EE, 128, 0, stream>>>(ef, ln1e_g, ln1e_b);
    ffn32<<<gN32, 512, 0, stream>>>(hf, Fh1_w, Fh1_b, Fh2_w, Fh2_b, NN);
    ln128<<<NN, 128, 0, stream>>>(hf, ln2h_g, ln2h_b);
    ffn32<<<gE32, 512, 0, stream>>>(ef, Fe1_w, Fe1_b, Fe2_w, Fe2_b, EE);
    ln128<<<EE, 128, 0, stream>>>(ef, ln2e_g, ln2e_b);

    // --- cross-attention to word embeddings ---
    gemm32<HID, false><<<gN32, 256, 0, stream>>>(hf, HID, ca_q, HID, nullptr, nullptr, Kb, NN, 0);
    gemm32<WDIM, true><<<gW32, 256, 0, stream>>>(word, WDIM, ca_k, HID, nullptr, nullptr, Vb, KW, 0);
    gemm32<WDIM, true><<<gW32, 256, 0, stream>>>(word, WDIM, ca_v, HID, nullptr, nullptr, vv, KW, 0);
    transpose_kT<<<(128 * 512 + 255) / 256, 256, 0, stream>>>(Vb, kT);
    cross_attn<<<gN16, 256, 0, stream>>>(Kb, kT, vv, logits, attv);

    gemm32<HID, false><<<gN32, 256, 0, stream>>>(attv, HID, ca_att, HID, nullptr, hf, nf, NN, 0);
    ln128<<<NN, 128, 0, stream>>>(nf, ca_ln1_g, ca_ln1_b);
    ffn32<<<gN32, 512, 0, stream>>>(nf, ca_f1, nullptr, ca_f2, nullptr, NN);
    ln128<<<NN, 128, 0, stream>>>(nf, ca_ln2_g, ca_ln2_b);
}
